// Attention_4758823764649
// MI455X (gfx1250) — matrix operand order repaired and checked by reading, not compiled, not run on hardware
//
#include <hip/hip_runtime.h>
#include <hip/hip_bf16.h>

#define DIM 768
#define NH 12
#define DH 64
#define BB 4
#define NN 1024
#define NT (BB * NN)          // 4096 tokens
#define QK_SCALE 0.125f       // 64^-0.5

typedef __attribute__((ext_vector_type(16))) _Float16 v16h;
typedef __attribute__((ext_vector_type(8)))  _Float16 v8h;
typedef __attribute__((ext_vector_type(8)))  float    v8f;
typedef __attribute__((ext_vector_type(4))) unsigned int u32x4;
typedef __attribute__((ext_vector_type(8))) int          i32x8;
typedef __attribute__((ext_vector_type(4))) int          i32x4;

// ---------------------------------------------------------------------------
// WMMA helper: D = A(16x32 f16) * B(32x16 f16) + C(16x16 f32)
// ---------------------------------------------------------------------------
__device__ __forceinline__ v8f wmma_f16(v16h a, v16h b, v8f c) {
  return __builtin_amdgcn_wmma_f32_16x16x32_f16(
      /*neg_a=*/false, a, /*neg_b=*/false, b,
      /*c_mod=*/(short)0, c, /*reuse_a=*/false, /*reuse_b=*/false);
}

// A-matrix fragment (16x32 f16), contiguous k order:
// lane r(0-15): M=r, K = {g*16..g*16+15} (g=lane>>4)
__device__ __forceinline__ v16h load_afrag(const _Float16* base, int ld, int row0, int k0) {
  int lane = threadIdx.x & 31;
  int r = lane & 15;
  int g = (lane >> 4) << 4;
  const _Float16* p = base + (size_t)(row0 + r) * ld + k0 + g;
  v8h lo = *(const v8h*)p;
  v8h hi = *(const v8h*)(p + 8);
  v16h o;
#pragma unroll
  for (int i = 0; i < 8; ++i) { o[i] = lo[i]; o[i + 8] = hi[i]; }
  return o;
}

// B-matrix fragment (32x16 f16): lane n holds column n; lanes 0-15 K=0..15,
// lanes 16-31 K=16..31.  Column n of B == contiguous row n of row-major [N x K].
__device__ __forceinline__ v16h load_bfrag(const _Float16* base, int ld, int col0, int k0) {
  int lane = threadIdx.x & 31;
  int n = lane & 15;
  int kg = (lane >> 4) << 4;
  return *(const v16h*)(base + (size_t)(col0 + n) * ld + k0 + kg);
}

// Same fragment loaders against an LDS tile
__device__ __forceinline__ v16h load_bfrag_lds(const _Float16* tile, int ld, int col0, int k0) {
  int lane = threadIdx.x & 31;
  int n = lane & 15;
  int kg = (lane >> 4) << 4;
  return *(const v16h*)(tile + (col0 + n) * ld + k0 + kg);
}
__device__ __forceinline__ v16h load_afrag_lds(const _Float16* tile) {
  int lane = threadIdx.x & 31;
  int r = lane & 15;
  int g = (lane >> 4) << 4;
  const _Float16* p = tile + r * 32 + g;
  v8h lo = *(const v8h*)p;
  v8h hi = *(const v8h*)(p + 8);
  v16h o;
#pragma unroll
  for (int i = 0; i < 8; ++i) { o[i] = lo[i]; o[i + 8] = hi[i]; }
  return o;
}

// ---------------------------------------------------------------------------
// Tensor Data Mover: 2D tile (f16 elements) global -> LDS.
// Descriptor bitfields per CDNA5 ISA 8.3-8.6.  Issued by one wave; EXEC ignored.
//   tile_x  : elements per row of the tile
//   tile_y  : rows of the tile
//   stride_x: row stride of the source tensor (elements)
// LDS result: tile_y rows x tile_x f16, contiguous (row stride = tile_x).
// ---------------------------------------------------------------------------
__device__ __forceinline__ void tdm_load_2d_f16(unsigned lds_off, const void* gptr,
                                                unsigned tile_x, unsigned tile_y,
                                                unsigned tensor_x, unsigned tensor_y,
                                                unsigned stride_x) {
  unsigned long long ga = (unsigned long long)(uintptr_t)gptr;
  u32x4 g0;
  g0[0] = 1u;                                            // count=1, user-mode
  g0[1] = lds_off;                                       // lds_addr (bytes)
  g0[2] = (unsigned)(ga & 0xFFFFFFFFu);                  // global_addr[31:0]
  g0[3] = (unsigned)((ga >> 32) & 0x01FFFFFFu)           // global_addr[56:32]
        | (2u << 30);                                    // type = 2 ("image")
  i32x8 g1;
  g1[0] = (int)(1u << 16);                               // data_size=1 (2B), wg_mask=0
  g1[1] = (int)((tensor_x & 0xFFFFu) << 16);             // tensor_dim0[15:0]
  g1[2] = (int)((tensor_x >> 16) | ((tensor_y & 0xFFFFu) << 16)); // dim0 hi | dim1 lo
  g1[3] = (int)((tensor_y >> 16) | (tile_x << 16));      // dim1 hi | tile_dim0
  g1[4] = (int)(tile_y & 0xFFFFu);                       // tile_dim1 | tile_dim2=0
  g1[5] = (int)stride_x;                                 // tensor_dim0_stride[31:0]
  g1[6] = 0;                                             // stride hi | dim1_stride lo
  g1[7] = 0;
  i32x4 g2 = {0, 0, 0, 0};
  i32x4 g3 = {0, 0, 0, 0};
#if __clang_major__ >= 23
  i32x8 g4 = {0, 0, 0, 0, 0, 0, 0, 0};
  __builtin_amdgcn_tensor_load_to_lds(g0, g1, g2, g3, g4, 0);
#else
  __builtin_amdgcn_tensor_load_to_lds(g0, g1, g2, g3, 0);
#endif
}

__device__ __forceinline__ unsigned lds_offset_u32(const void* p) {
  // generic pointer to LDS carries the LDS byte offset in its low 32 bits
  return (unsigned)(uintptr_t)p;
}

// ---------------------------------------------------------------------------
// Kernel 1: pre-norm (mean, std ddof=1) + f16 conversion. One block per row.
// ---------------------------------------------------------------------------
__global__ __launch_bounds__(256) void ln_kernel(const float* __restrict__ x,
                                                 _Float16* __restrict__ xn) {
  __shared__ float s_sum[256];
  __shared__ float s_sq[256];
  int row = blockIdx.x;
  const float* xr = x + (size_t)row * DIM;
  float s = 0.f, ss = 0.f;
  for (int i = threadIdx.x; i < DIM; i += 256) {
    float v = xr[i];
    s += v; ss += v * v;
  }
  s_sum[threadIdx.x] = s; s_sq[threadIdx.x] = ss;
  __syncthreads();
  for (int off = 128; off > 0; off >>= 1) {
    if (threadIdx.x < off) {
      s_sum[threadIdx.x] += s_sum[threadIdx.x + off];
      s_sq[threadIdx.x]  += s_sq[threadIdx.x + off];
    }
    __syncthreads();
  }
  float mean = s_sum[0] * (1.0f / DIM);
  float var  = fmaxf((s_sq[0] - mean * mean * (float)DIM) * (1.0f / (DIM - 1)), 1e-12f);
  float rstd = rsqrtf(var);
  for (int i = threadIdx.x; i < DIM; i += 256)
    xn[(size_t)row * DIM + i] = (_Float16)((xr[i] - mean) * rstd);
}

// ---------------------------------------------------------------------------
// Kernel 2: f32 -> f16 conversion (weights)
// ---------------------------------------------------------------------------
__global__ __launch_bounds__(256) void cvt_f32_f16(const float* __restrict__ src,
                                                   _Float16* __restrict__ dst, int n) {
  int i = blockIdx.x * 256 + threadIdx.x;
  if (i < n) dst[i] = (_Float16)src[i];
}

// ---------------------------------------------------------------------------
// Kernel 3: QKV GEMM, 32x32 register tile per wave (2 A-frags x 2 B-frags ->
// 4 WMMAs per 32-deep k-step; halves vmem per wmma vs 16x16 tiling).
// mode 0: Q (scaled, [B,H,N,Dh])  mode 1: K ([B,H,N,Dh])  mode 2: V^T ([B,H,Dh,N])
// ---------------------------------------------------------------------------
__global__ __launch_bounds__(256) void gemm_qkv(const _Float16* __restrict__ A,
                                                const _Float16* __restrict__ W,
                                                const float* __restrict__ bias,
                                                _Float16* __restrict__ out, int mode) {
  int wave = threadIdx.x >> 5;
  int tile = blockIdx.x * 8 + wave;           // 0 .. 3071
  int tn = tile % (DIM / 32);                 // 0..23
  int tm = tile / (DIM / 32);                 // 0..127
  int lane = threadIdx.x & 31;
  int m0 = tm * 32, n0 = tn * 32;

  v8f c[2][2] = {};
#pragma unroll 2
  for (int k0 = 0; k0 < DIM; k0 += 32) {
    v16h a0 = load_afrag(A, DIM, m0,      k0);
    v16h a1 = load_afrag(A, DIM, m0 + 16, k0);
    v16h b0 = load_bfrag(W, DIM, n0,      k0);
    v16h b1 = load_bfrag(W, DIM, n0 + 16, k0);
    c[0][0] = wmma_f16(a0, b0, c[0][0]);
    c[0][1] = wmma_f16(a0, b1, c[0][1]);
    c[1][0] = wmma_f16(a1, b0, c[1][0]);
    c[1][1] = wmma_f16(a1, b1, c[1][1]);
  }

  int half = lane >> 4;
#pragma unroll
  for (int j = 0; j < 2; ++j) {
    int col = n0 + j * 16 + (lane & 15);
    float bv = bias[col];
    int h = col >> 6, d = col & 63;
#pragma unroll
    for (int i = 0; i < 2; ++i) {
#pragma unroll
      for (int v = 0; v < 8; ++v) {
        int t = m0 + i * 16 + v + 8 * half;
        int bidx = t >> 10, tok = t & 1023;
        float val = c[i][j][v] + bv;
        if (mode == 0)
          out[(((size_t)bidx * NH + h) * NN + tok) * DH + d] = (_Float16)(val * QK_SCALE);
        else if (mode == 1)
          out[(((size_t)bidx * NH + h) * NN + tok) * DH + d] = (_Float16)val;
        else
          out[(((size_t)bidx * NH + h) * DH + d) * NN + tok] = (_Float16)val;
      }
    }
  }
}

// ---------------------------------------------------------------------------
// Kernel 4: flash attention with TDM-staged, double-buffered K/V chunk tiles.
// Block = 8 waves x 16 query rows of one (b,h).  Per 32-key chunk:
//   TDM: K tile 32x64 f16, V^T tile 64x32 f16 -> LDS (shared by all 8 waves)
//   each wave: 4 WMMAs for S, online softmax, P via LDS transpose, 4 WMMAs PV
// ---------------------------------------------------------------------------
__global__ __launch_bounds__(256) void attn_kernel(const _Float16* __restrict__ Q,
                                                   const _Float16* __restrict__ K,
                                                   const _Float16* __restrict__ Vt,
                                                   _Float16* __restrict__ AO) {
  __shared__ __align__(16) _Float16 ktile[2][32 * DH];   // 2 x 4KB
  __shared__ __align__(16) _Float16 vtile[2][DH * 32];   // 2 x 4KB
  __shared__ __align__(16) _Float16 pbuf[8][16 * 32];    // 8KB: per-wave P tile

  int wave = threadIdx.x >> 5;
  int lane = threadIdx.x & 31;
  int bh = blockIdx.x >> 3;                              // 0..47
  int q0 = (blockIdx.x & 7) * 128 + wave * 16;
  const _Float16* Qh = Q  + (size_t)bh * NN * DH;
  const _Float16* Kh = K  + (size_t)bh * NN * DH;
  const _Float16* Vh = Vt + (size_t)bh * DH * NN;

  v16h qa0 = load_afrag(Qh, DH, q0, 0);
  v16h qa1 = load_afrag(Qh, DH, q0, 32);

  float M[8], L[8];
  v8f acc[4] = {};
#pragma unroll
  for (int v = 0; v < 8; ++v) { M[v] = -1e30f; L[v] = 0.f; }

  _Float16* pw = &pbuf[wave][0];
  int coln = lane & 15;
  const int NCHUNK = NN / 32;                            // 32

  // prologue: stage chunk 0
  if (wave == 0) {
    tdm_load_2d_f16(lds_offset_u32(&ktile[0][0]), Kh, DH, 32, DH, NN, DH);
    tdm_load_2d_f16(lds_offset_u32(&vtile[0][0]), Vh, 32, DH, NN, DH, NN);
  }

  for (int i = 0; i < NCHUNK; ++i) {
    int cur = i & 1, nxt = cur ^ 1;
    __syncthreads();   // prior readers of buf[nxt] done before TDM overwrites it
    if (wave == 0) {
      if (i + 1 < NCHUNK) {
        int kc1 = (i + 1) * 32;
        tdm_load_2d_f16(lds_offset_u32(&ktile[nxt][0]), Kh + (size_t)kc1 * DH,
                        DH, 32, DH, NN, DH);
        tdm_load_2d_f16(lds_offset_u32(&vtile[nxt][0]), Vh + kc1,
                        32, DH, NN, DH, NN);
        __builtin_amdgcn_s_wait_tensorcnt(2);  // chunk i landed; i+1 in flight
      } else {
        __builtin_amdgcn_s_wait_tensorcnt(0);
      }
    }
    __syncthreads();   // publish chunk i to all waves

    const _Float16* kt = &ktile[cur][0];
    const _Float16* vt = &vtile[cur][0];

    // S tiles for local keys [0,16) and [16,32); Dh=64 => 2 k-steps each
    v8f s0 = {}, s1 = {};
    s0 = wmma_f16(qa0, load_bfrag_lds(kt, DH, 0,  0),  s0);
    s0 = wmma_f16(qa1, load_bfrag_lds(kt, DH, 0,  32), s0);
    s1 = wmma_f16(qa0, load_bfrag_lds(kt, DH, 16, 0),  s1);
    s1 = wmma_f16(qa1, load_bfrag_lds(kt, DH, 16, 32), s1);

    // online softmax (row m = v + 8*(lane>>4); reductions within half-wave)
#pragma unroll
    for (int v = 0; v < 8; ++v) {
      float mx = fmaxf(s0[v], s1[v]);
#pragma unroll
      for (int msk = 8; msk > 0; msk >>= 1)
        mx = fmaxf(mx, __shfl_xor(mx, msk, 16));
      float newM = fmaxf(M[v], mx);
      float ef = __expf(M[v] - newM);
      M[v] = newM;
      float p0 = __expf(s0[v] - newM);
      float p1 = __expf(s1[v] - newM);
      float rs = p0 + p1;
#pragma unroll
      for (int msk = 8; msk > 0; msk >>= 1)
        rs += __shfl_xor(rs, msk, 16);
      L[v] = L[v] * ef + rs;
#pragma unroll
      for (int dt = 0; dt < 4; ++dt) acc[dt][v] *= ef;
      int m = v + 8 * (lane >> 4);
      pw[m * 32 + coln]      = (_Float16)p0;
      pw[m * 32 + 16 + coln] = (_Float16)p1;
    }

    // P (A-operand) from per-wave LDS tile; V^T rows from shared LDS tile
    v16h pa = load_afrag_lds(pw);
#pragma unroll
    for (int dt = 0; dt < 4; ++dt) {
      v16h vb = load_bfrag_lds(vt, 32, dt * 16, 0);
      acc[dt] = wmma_f16(pa, vb, acc[dt]);
    }
  }

  // epilogue: O /= L, scatter to [B, N, DIM] f16
  int b = bh / NH, h = bh % NH;
  int half = lane >> 4;
#pragma unroll
  for (int v = 0; v < 8; ++v) {
    float inv = 1.0f / L[v];
    int tok = q0 + v + 8 * half;
    size_t rowoff = ((size_t)b * NN + tok) * DIM + h * DH;
#pragma unroll
    for (int dt = 0; dt < 4; ++dt)
      AO[rowoff + dt * 16 + coln] = (_Float16)(acc[dt][v] * inv);
  }
}

// ---------------------------------------------------------------------------
// Kernel 5: output projection + bias + residual (f32 out), 32x32 wave tile
// ---------------------------------------------------------------------------
__global__ __launch_bounds__(256) void gemm_proj(const _Float16* __restrict__ A,
                                                 const _Float16* __restrict__ W,
                                                 const float* __restrict__ bias,
                                                 const float* __restrict__ x,
                                                 float* __restrict__ out) {
  int wave = threadIdx.x >> 5;
  int tile = blockIdx.x * 8 + wave;
  int tn = tile % (DIM / 32);
  int tm = tile / (DIM / 32);
  int lane = threadIdx.x & 31;
  int m0 = tm * 32, n0 = tn * 32;

  v8f c[2][2] = {};
#pragma unroll 2
  for (int k0 = 0; k0 < DIM; k0 += 32) {
    v16h a0 = load_afrag(A, DIM, m0,      k0);
    v16h a1 = load_afrag(A, DIM, m0 + 16, k0);
    v16h b0 = load_bfrag(W, DIM, n0,      k0);
    v16h b1 = load_bfrag(W, DIM, n0 + 16, k0);
    c[0][0] = wmma_f16(a0, b0, c[0][0]);
    c[0][1] = wmma_f16(a0, b1, c[0][1]);
    c[1][0] = wmma_f16(a1, b0, c[1][0]);
    c[1][1] = wmma_f16(a1, b1, c[1][1]);
  }

  int half = lane >> 4;
#pragma unroll
  for (int j = 0; j < 2; ++j) {
    int col = n0 + j * 16 + (lane & 15);
    float bv = bias[col];
#pragma unroll
    for (int i = 0; i < 2; ++i) {
#pragma unroll
      for (int v = 0; v < 8; ++v) {
        int t = m0 + i * 16 + v + 8 * half;
        size_t idx = (size_t)t * DIM + col;
        out[idx] = c[i][j][v] + bv + x[idx];
      }
    }
  }
}

// ---------------------------------------------------------------------------
// Host launcher
// ---------------------------------------------------------------------------
extern "C" void kernel_launch(void* const* d_in, const int* in_sizes, int n_in,
                              void* d_out, int out_size, void* d_ws, size_t ws_size,
                              hipStream_t stream) {
  const float* x   = (const float*)d_in[0];
  const float* qw  = (const float*)d_in[1];
  const float* kw  = (const float*)d_in[2];
  const float* vw  = (const float*)d_in[3];
  const float* qb  = (const float*)d_in[4];
  const float* kb  = (const float*)d_in[5];
  const float* vb  = (const float*)d_in[6];
  const float* pjw = (const float*)d_in[7];
  const float* pjb = (const float*)d_in[8];
  float* out = (float*)d_out;

  _Float16* ws = (_Float16*)d_ws;
  size_t o = 0;
  _Float16* xn = ws + o; o += (size_t)NT * DIM;
  _Float16* wq = ws + o; o += (size_t)DIM * DIM;
  _Float16* wk = ws + o; o += (size_t)DIM * DIM;
  _Float16* wv = ws + o; o += (size_t)DIM * DIM;
  _Float16* wp = ws + o; o += (size_t)DIM * DIM;
  _Float16* Qs = ws + o; o += (size_t)NT * DIM;     // [B,H,N,Dh]
  _Float16* Ks = ws + o; o += (size_t)NT * DIM;     // [B,H,N,Dh]
  _Float16* Vt = ws + o; o += (size_t)NT * DIM;     // [B,H,Dh,N]
  _Float16* AO = ws + o; o += (size_t)NT * DIM;     // [B,N,DIM]

  ln_kernel<<<NT, 256, 0, stream>>>(x, xn);

  const int wn = DIM * DIM;
  const int cvb = (wn + 255) / 256;
  cvt_f32_f16<<<cvb, 256, 0, stream>>>(qw,  wq, wn);
  cvt_f32_f16<<<cvb, 256, 0, stream>>>(kw,  wk, wn);
  cvt_f32_f16<<<cvb, 256, 0, stream>>>(vw,  wv, wn);
  cvt_f32_f16<<<cvb, 256, 0, stream>>>(pjw, wp, wn);

  // 3072 wave-tiles of 32x32 / 8 waves = 384 blocks
  gemm_qkv<<<384, 256, 0, stream>>>(xn, wq, qb, Qs, 0);
  gemm_qkv<<<384, 256, 0, stream>>>(xn, wk, kb, Ks, 1);
  gemm_qkv<<<384, 256, 0, stream>>>(xn, wv, vb, Vt, 2);

  // 48 (b,h) * 8 q-blocks = 384 blocks, 8 waves x 16 q-rows
  attn_kernel<<<384, 256, 0, stream>>>(Qs, Ks, Vt, AO);

  gemm_proj<<<384, 256, 0, stream>>>(AO, wp, pjb, x, out);
}
